// NNFOwithBayesianJumps_69784628626156
// MI455X (gfx1250) — matrix operand order repaired and checked by reading, not compiled, not run on hardware
//
#include <hip/hip_runtime.h>
#include <hip/hip_bf16.h>

typedef __attribute__((ext_vector_type(4)))  unsigned int u4;
typedef __attribute__((ext_vector_type(4)))  int          i4;
typedef __attribute__((ext_vector_type(16))) __bf16       v16bf;
typedef __attribute__((ext_vector_type(8)))  float        v8f;

typedef __attribute__((address_space(1))) i4 gi4;   // global-space v4i
typedef __attribute__((address_space(3))) i4 li4;   // LDS-space v4i

#define T_STEPS 256
#define BATCH   1024
#define DIN     64
#define HID     768
#define PHID    512
#define PREP    10
#define COVSZ   32
#define COVHID  128
#define DT_EULER 0.05f
#define MIX      1.0f
#define VAR_EPS  1e-6f

#if __has_builtin(__builtin_amdgcn_global_load_async_to_lds_b128)
#define HAVE_ASYNC_LDS 1
#else
#define HAVE_ASYNC_LDS 0
#endif

union FragBF { v16bf v; u4 q[2]; };

__device__ __forceinline__ float sigmoidf(float x) { return 1.0f / (1.0f + __expf(-x)); }

__device__ __forceinline__ void stage16B(const __bf16* gsrc, __bf16* ldst) {
#if HAVE_ASYNC_LDS
    __builtin_amdgcn_global_load_async_to_lds_b128((gi4*)gsrc, (li4*)ldst, 0, 0);
#else
    *(u4*)ldst = *(const u4*)gsrc;
#endif
}

// wait until at most `pend` async global->LDS ops remain outstanding
__device__ __forceinline__ void stage_wait_all() {
#if HAVE_ASYNC_LDS
#if __has_builtin(__builtin_amdgcn_s_wait_asynccnt)
    __builtin_amdgcn_s_wait_asynccnt(0);
#else
    asm volatile("s_wait_asynccnt 0" ::: "memory");
#endif
#endif
}
__device__ __forceinline__ void stage_wait_keep4() {
#if HAVE_ASYNC_LDS
#if __has_builtin(__builtin_amdgcn_s_wait_asynccnt)
    __builtin_amdgcn_s_wait_asynccnt(4);
#else
    asm volatile("s_wait_asynccnt 4" ::: "memory");
#endif
#endif
}

// ---------------- fp32 -> bf16 convert ----------------
__global__ __launch_bounds__(256) void k_f2bf(const float* __restrict__ in,
                                              __bf16* __restrict__ out, int n) {
    int i = blockIdx.x * 256 + threadIdx.x;
    if (i < n) out[i] = (__bf16)in[i];
}

__global__ void k_zero(float* p) {
    if (blockIdx.x == 0 && threadIdx.x == 0) p[0] = 0.0f;
}

// ---------------- h0 init: covariates map ----------------
__global__ __launch_bounds__(256) void k_cov1(const float* __restrict__ cov,
                                              const float* __restrict__ Wc1,
                                              const float* __restrict__ bc1,
                                              float* __restrict__ tmp) {
    int idx = blockIdx.x * 256 + threadIdx.x;          // B*COVHID
    if (idx >= BATCH * COVHID) return;
    int b = idx >> 7, j = idx & 127;
    float s = bc1[j];
    #pragma unroll
    for (int c = 0; c < COVSZ; ++c) s += cov[b * COVSZ + c] * Wc1[j * COVSZ + c];
    tmp[idx] = fmaxf(s, 0.0f);
}

__global__ __launch_bounds__(256) void k_cov2(const float* __restrict__ tmp,
                                              const float* __restrict__ Wc2,
                                              const float* __restrict__ bc2,
                                              float* __restrict__ h,
                                              __bf16* __restrict__ hbf) {
    int idx = blockIdx.x * 256 + threadIdx.x;          // B*HID
    if (idx >= BATCH * HID) return;
    int b = idx / HID, i = idx % HID;
    float s = bc2[i];
    for (int j = 0; j < COVHID; ++j) s += tmp[b * COVHID + j] * Wc2[i * COVHID + j];
    float v = tanhf(s);
    h[idx] = v;
    hbf[idx] = (__bf16)v;
}

// ---------------- WMMA GEMM with fused epilogues ----------------
// out[m,n] = epilogue( A1[m,:K1] . W1[n,:K1] + A2[m,:K2] . W2[n,:K2] + bias[n] )
// mode 0: identity          -> outF (opt), outB (opt)
// mode 1: relu              -> outB (bf16), outF (opt)
// mode 2: z = sigmoid(pre)  -> outF = z, outB = bf16(z * auxH)
// mode 3: n = tanh(pre)     -> hN = auxH + DT*(1-auxZ)*(n-auxH) -> outF, outB
#define BM 128
#define BN 128
#define BK 32
#define LDT 40   // padded LDS row stride (elements) -> conflict-free 16B reads

__global__ __launch_bounds__(256) void k_gemm(
    const __bf16* __restrict__ A1, const __bf16* __restrict__ W1, int K1,
    const __bf16* __restrict__ A2, const __bf16* __restrict__ W2, int K2,
    const float* __restrict__ bias,
    float* __restrict__ outF, __bf16* __restrict__ outB,
    const float* __restrict__ auxH, const float* __restrict__ auxZ,
    int N, int mode)
{
    __shared__ __bf16 sA[2][BM * LDT];
    __shared__ __bf16 sW[2][BN * LDT];

    const int tid  = threadIdx.x;
    const int lane = tid & 31;
    const int wv   = tid >> 5;                  // 8 waves
    const int rowBase = blockIdx.y * BM;
    const int colBase = blockIdx.x * BN;

    const int nt1    = K1 / BK;
    const int nt2    = (A2 != nullptr) ? (K2 / BK) : 0;
    const int ntiles = nt1 + nt2;

    // stage tile t into LDS buffer (t&1): exactly 4 async 16B ops per thread
    auto stage_tile = [&](int t) {
        const int buf = t & 1;
        const __bf16* Ap; const __bf16* Wp; int K; int k0;
        if (t < nt1) { Ap = A1; Wp = W1; K = K1; k0 = t * BK; }
        else         { Ap = A2; Wp = W2; K = K2; k0 = (t - nt1) * BK; }
        #pragma unroll
        for (int i = tid; i < (BM * BK / 8); i += 256) {   // 2 iterations
            int r = i >> 2, s = i & 3;
            stage16B(&Ap[(size_t)(rowBase + r) * K + k0 + s * 8],
                     &sA[buf][r * LDT + s * 8]);
        }
        #pragma unroll
        for (int i = tid; i < (BN * BK / 8); i += 256) {   // 2 iterations
            int r = i >> 2, s = i & 3;
            stage16B(&Wp[(size_t)(colBase + r) * K + k0 + s * 8],
                     &sW[buf][r * LDT + s * 8]);
        }
    };

    v8f acc[8] = {};

    stage_tile(0);
    for (int t = 0; t < ntiles; ++t) {
        if (t + 1 < ntiles) {
            stage_tile(t + 1);       // prefetch next tile into other buffer
            stage_wait_keep4();      // tile t complete; next tile (4 ops) in flight
        } else {
            stage_wait_all();
        }
        __syncthreads();

        const int buf = t & 1;
        // A fragment (16x32 bf16): lane<16 -> K{0..7,16..23}, lane>=16 -> K{8..15,24..31}
        const int arow = wv * 16 + (lane & 15);
        const int kb   = (lane < 16) ? 0 : 8;
        FragBF a;
        a.q[0] = *(const u4*)&sA[buf][arow * LDT + kb];
        a.q[1] = *(const u4*)&sA[buf][arow * LDT + kb + 16];

        // load all 8 B fragments first (distinct regs -> ds_loads can run ahead)
        const int kb2 = (lane < 16) ? 0 : 8;
        FragBF bfr[8];
        #pragma unroll
        for (int c = 0; c < 8; ++c) {
            const int wrow = c * 16 + (lane & 15);
            bfr[c].q[0] = *(const u4*)&sW[buf][wrow * LDT + kb2];
            bfr[c].q[1] = *(const u4*)&sW[buf][wrow * LDT + kb2 + 16];
        }
        #pragma unroll
        for (int c = 0; c < 8; ++c) {
            acc[c] = __builtin_amdgcn_wmma_f32_16x16x32_bf16(
                false, a.v, false, bfr[c].v, (short)0, acc[c], false, false);
        }
        __syncthreads();
    }

    // epilogue: C layout -> lane<16: (M=r, N=lane); lane>=16: (M=8+r, N=lane-16)
    #pragma unroll
    for (int c = 0; c < 8; ++c) {
        const int col = colBase + c * 16 + (lane & 15);
        const float bv = bias ? bias[col] : 0.0f;
        #pragma unroll
        for (int r = 0; r < 8; ++r) {
            const int row = rowBase + wv * 16 + ((lane < 16) ? r : r + 8);
            const size_t idx = (size_t)row * N + col;
            float v = acc[c][r] + bv;
            if (mode == 0) {
                if (outF) outF[idx] = v;
                if (outB) outB[idx] = (__bf16)v;
            } else if (mode == 1) {
                v = fmaxf(v, 0.0f);
                if (outF) outF[idx] = v;
                outB[idx] = (__bf16)v;
            } else if (mode == 2) {
                const float z = sigmoidf(v);
                outF[idx] = z;
                outB[idx] = (__bf16)(z * auxH[idx]);
            } else {
                const float n  = tanhf(v);
                const float h  = auxH[idx];
                const float z  = auxZ[idx];
                const float hn = h + DT_EULER * (1.0f - z) * (n - h);
                outF[idx] = hn;
                outB[idx] = (__bf16)hn;
            }
        }
    }
}

// ---------------- NLL (+ optional prep of gi) ----------------
__global__ __launch_bounds__(256) void k_nll(
    const float* __restrict__ p, const float* __restrict__ Xt,
    const float* __restrict__ Mt, const int* __restrict__ om,
    const float* __restrict__ w_prep, const float* __restrict__ bias_prep,
    __bf16* __restrict__ gi_bf, float* __restrict__ lossAcc,
    float scale, int doPrep)
{
    const int tid = threadIdx.x;
    const int idx = blockIdx.x * 256 + tid;      // B*DIN threads
    const int b = idx >> 6, d = idx & 63;

    const float mean = p[b * (2 * DIN) + d];
    const float var  = p[b * (2 * DIN) + DIN + d];
    const float av   = fabsf(var) + VAR_EPS;
    const float x    = Xt[idx];
    const float mm   = Mt[idx] * (om[b] ? 1.0f : 0.0f);
    const float err  = (x - mean) * rsqrtf(av);
    const float l    = 0.5f * (err * err + __logf(av)) * mm * scale;

    __shared__ float red[256];
    red[tid] = l;
    __syncthreads();
    #pragma unroll
    for (int s = 128; s > 0; s >>= 1) {
        if (tid < s) red[tid] += red[tid + s];
        __syncthreads();
    }
    if (tid == 0) atomicAdd(lossAcc, red[0]);

    if (doPrep) {
        const float* w = w_prep + d * 4 * PREP;     // (d, f, p)
        #pragma unroll
        for (int pp = 0; pp < PREP; ++pp) {
            float a = x * w[pp] + mean * w[PREP + pp] + av * w[2 * PREP + pp]
                    + err * w[3 * PREP + pp] + bias_prep[d * PREP + pp];
            a = fmaxf(a, 0.0f) * mm;
            gi_bf[(size_t)b * (DIN * PREP) + d * PREP + pp] = (__bf16)a;
        }
    }
}

// ---------------- GRU gates + Bayesian jump ----------------
__global__ __launch_bounds__(256) void k_gate(
    const float* __restrict__ Gi, const float* __restrict__ Gh,
    const int* __restrict__ om, float* __restrict__ h, __bf16* __restrict__ hbf)
{
    const int idx = blockIdx.x * 256 + threadIdx.x;   // B*HID
    if (idx >= BATCH * HID) return;
    const int b = idx / HID, i = idx % HID;
    const size_t base = (size_t)b * (3 * HID);
    const float r  = sigmoidf(Gi[base + i]            + Gh[base + i]);
    const float z  = sigmoidf(Gi[base + HID + i]      + Gh[base + HID + i]);
    const float n  = tanhf  (Gi[base + 2 * HID + i] + r * Gh[base + 2 * HID + i]);
    const float hv = h[idx];
    const float hnew = (1.0f - z) * n + z * hv;
    const float out = om[b] ? hnew : hv;
    h[idx] = out;
    hbf[idx] = (__bf16)out;
}

__global__ __launch_bounds__(256) void k_final(const float* __restrict__ h,
                                               const float* __restrict__ loss,
                                               float* __restrict__ out) {
    const int i = blockIdx.x * 256 + threadIdx.x;
    if (i < BATCH * HID) out[i] = h[i];
    if (i == BATCH * HID) out[BATCH * HID] = loss[0];
}

// =====================================================================
extern "C" void kernel_launch(void* const* d_in, const int* in_sizes, int n_in,
                              void* d_out, int out_size, void* d_ws, size_t ws_size,
                              hipStream_t stream)
{
    const float* X    = (const float*)d_in[0];
    const float* Mmat = (const float*)d_in[1];
    const int*   obs  = (const int*)  d_in[2];
    const float* cov  = (const float*)d_in[3];
    const float* Wxz  = (const float*)d_in[4];  const float* bxz = (const float*)d_in[5];
    const float* Wxn  = (const float*)d_in[6];  const float* bxn = (const float*)d_in[7];
    const float* Whz  = (const float*)d_in[8];  const float* Whn = (const float*)d_in[9];
    const float* Wp1  = (const float*)d_in[10]; const float* bp1 = (const float*)d_in[11];
    const float* Wp2  = (const float*)d_in[12]; const float* bp2 = (const float*)d_in[13];
    const float* Wc1  = (const float*)d_in[14]; const float* bc1 = (const float*)d_in[15];
    const float* Wc2  = (const float*)d_in[16]; const float* bc2 = (const float*)d_in[17];
    const float* Wih  = (const float*)d_in[18]; const float* Whh = (const float*)d_in[19];
    const float* bih  = (const float*)d_in[20]; const float* bhh = (const float*)d_in[21];
    const float* w_prep    = (const float*)d_in[22];
    const float* bias_prep = (const float*)d_in[23];

    size_t off = 0;
    auto alloc = [&](size_t bytes) -> void* {
        off = (off + 255) & ~(size_t)255;
        void* p = (char*)d_ws + off;
        off += bytes;
        return p;
    };

    // bf16 weight copies
    __bf16* wBxz = (__bf16*)alloc((size_t)HID * 2 * DIN * 2);
    __bf16* wBxn = (__bf16*)alloc((size_t)HID * 2 * DIN * 2);
    __bf16* wBhz = (__bf16*)alloc((size_t)HID * HID * 2);
    __bf16* wBhn = (__bf16*)alloc((size_t)HID * HID * 2);
    __bf16* wBp1 = (__bf16*)alloc((size_t)PHID * HID * 2);
    __bf16* wBp2 = (__bf16*)alloc((size_t)2 * DIN * PHID * 2);
    __bf16* wBih = (__bf16*)alloc((size_t)3 * HID * DIN * PREP * 2);
    __bf16* wBhh = (__bf16*)alloc((size_t)3 * HID * HID * 2);

    // activations
    float*  hA    = (float*) alloc((size_t)BATCH * HID * 4);
    __bf16* hAbf  = (__bf16*)alloc((size_t)BATCH * HID * 2);
    float*  hB    = (float*) alloc((size_t)BATCH * HID * 4);
    __bf16* hBbf  = (__bf16*)alloc((size_t)BATCH * HID * 2);
    __bf16* r1bf  = (__bf16*)alloc((size_t)BATCH * PHID * 2);
    __bf16* pbf   = (__bf16*)alloc((size_t)BATCH * 2 * DIN * 2);
    float*  p2f   = (float*) alloc((size_t)BATCH * 2 * DIN * 4);
    float*  zf    = (float*) alloc((size_t)BATCH * HID * 4);
    __bf16* zhbf  = (__bf16*)alloc((size_t)BATCH * HID * 2);
    __bf16* gibf  = (__bf16*)alloc((size_t)BATCH * DIN * PREP * 2);
    float*  Gi    = (float*) alloc((size_t)BATCH * 3 * HID * 4);
    float*  Gh    = (float*) alloc((size_t)BATCH * 3 * HID * 4);
    float*  tmpc  = (float*) alloc((size_t)BATCH * COVHID * 4);
    float*  lossA = (float*) alloc(256);

    auto conv = [&](const float* src, __bf16* dst, int n) {
        k_f2bf<<<(n + 255) / 256, 256, 0, stream>>>(src, dst, n);
    };
    conv(Wxz, wBxz, HID * 2 * DIN);
    conv(Wxn, wBxn, HID * 2 * DIN);
    conv(Whz, wBhz, HID * HID);
    conv(Whn, wBhn, HID * HID);
    conv(Wp1, wBp1, PHID * HID);
    conv(Wp2, wBp2, 2 * DIN * PHID);
    conv(Wih, wBih, 3 * HID * DIN * PREP);
    conv(Whh, wBhh, 3 * HID * HID);

    k_zero<<<1, 1, 0, stream>>>(lossA);
    k_cov1<<<(BATCH * COVHID + 255) / 256, 256, 0, stream>>>(cov, Wc1, bc1, tmpc);
    k_cov2<<<(BATCH * HID + 255) / 256, 256, 0, stream>>>(tmpc, Wc2, bc2, hA, hAbf);

    auto gemm = [&](const __bf16* A1, const __bf16* W1, int K1,
                    const __bf16* A2, const __bf16* W2, int K2,
                    const float* bias, float* outF, __bf16* outB,
                    const float* auxH, const float* auxZ, int N, int mode) {
        dim3 grid(N / BN, BATCH / BM);
        k_gemm<<<grid, 256, 0, stream>>>(A1, W1, K1, A2, W2, K2, bias,
                                         outF, outB, auxH, auxZ, N, mode);
    };

    float*  hF  = hA;  __bf16* hBf  = hAbf;    // current h
    float*  hNF = hB;  __bf16* hNBf = hBbf;    // next h (ODE output)

    const int nllBlocks  = (BATCH * DIN) / 256;
    const int gateBlocks = (BATCH * HID + 255) / 256;

    for (int t = 0; t < T_STEPS; ++t) {
        const float* Xt = X    + (size_t)t * BATCH * DIN;
        const float* Mt = Mmat + (size_t)t * BATCH * DIN;
        const int*   om = obs  + (size_t)t * BATCH;

        // p = p_model(h)  (bf16 only)
        gemm(hBf, wBp1, HID, nullptr, nullptr, 0, bp1, nullptr, r1bf, nullptr, nullptr, PHID, 1);
        gemm(r1bf, wBp2, PHID, nullptr, nullptr, 0, bp2, nullptr, pbf, nullptr, nullptr, 2 * DIN, 0);

        // z = sigmoid(p@Wxz^T + bxz + h@Whz^T); zh_bf = bf16(z*h)
        gemm(pbf, wBxz, 2 * DIN, hBf, wBhz, HID, bxz, zf, zhbf, hF, nullptr, HID, 2);
        // n = tanh(p@Wxn^T + bxn + (z*h)@Whn^T); hN = h + DT*(1-z)*(n-h)
        gemm(pbf, wBxn, 2 * DIN, zhbf, wBhn, HID, bxn, hNF, hNBf, hF, zf, HID, 3);

        // p2 = p_model(hN)  (f32 for NLL/prep)
        gemm(hNBf, wBp1, HID, nullptr, nullptr, 0, bp1, nullptr, r1bf, nullptr, nullptr, PHID, 1);
        gemm(r1bf, wBp2, PHID, nullptr, nullptr, 0, bp2, p2f, nullptr, nullptr, nullptr, 2 * DIN, 0);

        // loss_pre + gi prep
        k_nll<<<nllBlocks, 256, 0, stream>>>(p2f, Xt, Mt, om, w_prep, bias_prep,
                                             gibf, lossA, 1.0f, 1);

        // GRU cell gates
        gemm(gibf, wBih, DIN * PREP, nullptr, nullptr, 0, bih, Gi, nullptr, nullptr, nullptr, 3 * HID, 0);
        gemm(hNBf, wBhh, HID,        nullptr, nullptr, 0, bhh, Gh, nullptr, nullptr, nullptr, 3 * HID, 0);
        k_gate<<<gateBlocks, 256, 0, stream>>>(Gi, Gh, om, hNF, hNBf);

        // p3 = p_model(h) for loss_post
        gemm(hNBf, wBp1, HID, nullptr, nullptr, 0, bp1, nullptr, r1bf, nullptr, nullptr, PHID, 1);
        gemm(r1bf, wBp2, PHID, nullptr, nullptr, 0, bp2, p2f, nullptr, nullptr, nullptr, 2 * DIN, 0);
        k_nll<<<nllBlocks, 256, 0, stream>>>(p2f, Xt, Mt, om, w_prep, bias_prep,
                                             gibf, lossA, MIX, 0);

        // swap h buffers
        float*  tf = hF;  hF = hNF;   hNF = tf;
        __bf16* tb = hBf; hBf = hNBf; hNBf = tb;
    }

    k_final<<<(BATCH * HID + 256) / 256 + 1, 256, 0, stream>>>(hF, lossA, (float*)d_out);
}
